// PatchGCN_9869834846749
// MI455X (gfx1250) — compile-verified, matrix operand order checked by reading, not run on hardware
//
#include <hip/hip_runtime.h>
#include <hip/hip_bf16.h>
#include <cstdint>

// ---------------------------------------------------------------------------
// PatchGCN forward for MI455X (gfx1250, wave32).
// GEMMs: v_wmma_f32_16x16x32_bf16, A-tiles moved by the Tensor Data Mover
// (tensor_load_to_lds) with TDM zero-padding for the M overhang and TDM LDS
// padding (32dw interval, 4dw pad) to kill bank conflicts.
// Edge softmax-aggregation is the bandwidth bottleneck (~2GB of gathers and
// atomics @ 23.3TB/s vs ~165 GFLOP of matrix math) and stays coalesced b32.
// ---------------------------------------------------------------------------

typedef __bf16 bf16_t;
typedef __attribute__((ext_vector_type(16))) bf16_t v16bf;
typedef __attribute__((ext_vector_type(8)))  bf16_t v8bf;
typedef __attribute__((ext_vector_type(8)))  float  v8f;
typedef __attribute__((ext_vector_type(4)))  unsigned int v4u;
typedef __attribute__((ext_vector_type(8)))  int v8i;
typedef __attribute__((ext_vector_type(4)))  int v4i;

enum { ACT_NONE = 0, ACT_RELU = 1, ACT_TANH = 2, ACT_SIG = 3 };

// Issue a TDM load of a 32(x)*64(y) fp32 tile at `gptr` (row stride lda
// elements) into LDS at byte offset lds_off, with rows >= rows_left reading
// as zero and 4-dword LDS padding after every 32-dword row (-> 36dw stride).
__device__ __forceinline__ void tdm_load_tile_f32(
    const float* gptr, uint32_t lds_off, int lda, int rows_left)
{
    uint64_t ga = (uint64_t)(uintptr_t)gptr;
    v4u g0;
    g0[0] = 1u;                                    // count=1, user mode
    g0[1] = lds_off;                               // lds_addr (bytes)
    g0[2] = (uint32_t)ga;                          // global_addr[31:0]
    g0[3] = (uint32_t)(ga >> 32) | (2u << 30);     // addr[56:32] | type=2
    v8i g1;
    g1[0] = (int)((2u << 16)      // data_size = 4B
                | (1u << 20)      // pad_enable
                | (4u << 22)      // pad_interval: 32 dwords
                | (3u << 25));    // pad_amount : 4 dwords
    g1[1] = (int)((32u & 0xffffu) << 16);            // tensor_dim0 = 32 (lo16)
    g1[2] = (int)(((uint32_t)rows_left & 0xffffu) << 16); // td0 hi | td1 lo
    g1[3] = (int)((((uint32_t)rows_left >> 16) & 0xffffu) | (32u << 16)); // td1 hi | tile_dim0=32
    g1[4] = 64;                                      // tile_dim1=64, tile_dim2=0
    g1[5] = lda;                                     // tensor_dim0_stride lo32
    g1[6] = 0;                                       // stride hi | dim1_stride lo
    g1[7] = 0;
    v4i g2 = {0, 0, 0, 0}, g3 = {0, 0, 0, 0};        // 2D: groups 2/3 unused
    v8i gx = {0, 0, 0, 0, 0, 0, 0, 0};               // clang-23 6-arg form
    __builtin_amdgcn_tensor_load_to_lds(g0, g1, g2, g3, gx, 0);
}

// Build bf16 WMMA A-fragment from a padded fp32 LDS row (ISA 7.12.2 layout:
// per lane two contiguous 8-element k-runs: 8*hsel and 16+8*hsel).
__device__ __forceinline__ v16bf frag_a_from_f32(const float* ar, int hsel)
{
    const float4 c0 = *(const float4*)(ar + 8 * hsel);
    const float4 c1 = *(const float4*)(ar + 8 * hsel + 4);
    const float4 c2 = *(const float4*)(ar + 8 * hsel + 16);
    const float4 c3 = *(const float4*)(ar + 8 * hsel + 20);
    v16bf af;
    af[0]  = (bf16_t)c0.x; af[1]  = (bf16_t)c0.y; af[2]  = (bf16_t)c0.z; af[3]  = (bf16_t)c0.w;
    af[4]  = (bf16_t)c1.x; af[5]  = (bf16_t)c1.y; af[6]  = (bf16_t)c1.z; af[7]  = (bf16_t)c1.w;
    af[8]  = (bf16_t)c2.x; af[9]  = (bf16_t)c2.y; af[10] = (bf16_t)c2.z; af[11] = (bf16_t)c2.w;
    af[12] = (bf16_t)c3.x; af[13] = (bf16_t)c3.y; af[14] = (bf16_t)c3.z; af[15] = (bf16_t)c3.w;
    return af;
}

__device__ __forceinline__ v16bf frag_b_from_bf16(const bf16_t* br, int hsel)
{
    v8bf b0 = *(const v8bf*)(br + 8 * hsel);
    v8bf b1 = *(const v8bf*)(br + 16 + 8 * hsel);
    return __builtin_shufflevector(b0, b1, 0, 1, 2, 3, 4, 5, 6, 7,
                                   8, 9, 10, 11, 12, 13, 14, 15);
}

// ---------------- WMMA GEMM: C = act(A @ B + bias) -------------------------
// A: [M,K] fp32 (row stride lda), B: [K,Nn] fp32 row-major, C fp32 (ldc).
// Block tile 64x32, 8 waves (4x2), one 16x16x32 WMMA per wave per K-step.
// Requires K % 32 == 0, Nn % 32 == 0, 16B-aligned rows (lda, Nn mult of 4).
template <int ACT>
__global__ __launch_bounds__(256) void gemm_kernel(
    const float* __restrict__ A, const float* __restrict__ B,
    const float* __restrict__ bias, float* __restrict__ C,
    int M, int K, int Nn, int lda, int ldc)
{
    __shared__ float  Araw[64][36];    // TDM-padded fp32 A tile (36dw stride)
    __shared__ bf16_t Bst[32][40];     // transposed bf16 B tile [col][k]

    const int tid  = threadIdx.x;
    const int lane = tid & 31;
    const int wave = tid >> 5;
    const int wm = wave >> 1, wn = wave & 1;
    const int r16 = lane & 15, hsel = lane >> 4;
    const int m0 = blockIdx.y * 64;
    const int n0 = blockIdx.x * 32;
    const uint32_t a_lds = (uint32_t)(uintptr_t)(void*)&Araw[0][0];

    v8f acc = {0.f, 0.f, 0.f, 0.f, 0.f, 0.f, 0.f, 0.f};

    const int bk = tid >> 3;             // 0..31 (k within tile)
    const int bc = (tid & 7) * 4;        // 0..28 (col group)

    for (int k0 = 0; k0 < K; k0 += 32) {
        if (wave == 0)  // async A tile via Tensor Data Mover
            tdm_load_tile_f32(A + (size_t)m0 * lda + k0, a_lds, lda, M - m0);
        // stage B 32x32 -> bf16 transposed
        {
            const float4 v = *(const float4*)(B + (size_t)(k0 + bk) * Nn + n0 + bc);
            Bst[bc + 0][bk] = (bf16_t)v.x;
            Bst[bc + 1][bk] = (bf16_t)v.y;
            Bst[bc + 2][bk] = (bf16_t)v.z;
            Bst[bc + 3][bk] = (bf16_t)v.w;
        }
        if (wave == 0) __builtin_amdgcn_s_wait_tensorcnt(0);
        __syncthreads();

        v16bf af  = frag_a_from_f32(&Araw[wm * 16 + r16][0], hsel);
        v16bf bfv = frag_b_from_bf16(&Bst[wn * 16 + r16][0], hsel);
        acc = __builtin_amdgcn_wmma_f32_16x16x32_bf16(
            false, af, false, bfv, (short)0, acc, false, false);
        __syncthreads();
    }

    const int col = n0 + wn * 16 + r16;
    const float bv = bias[col];
#pragma unroll
    for (int i = 0; i < 8; ++i) {
        int row = m0 + wm * 16 + i + 8 * hsel;   // C/D layout: VGPR i -> M=i/i+8
        if (row < M) {
            float v = acc[i] + bv;
            if (ACT == ACT_RELU)      v = fmaxf(v, 0.f);
            else if (ACT == ACT_TANH) v = tanhf(v);
            else if (ACT == ACT_SIG)  v = 1.f / (1.f + __expf(-v));
            C[(size_t)row * ldc + col] = v;
        }
    }
}

// Sigmoid GEMM with fused gated-attention dot:
//   s[row] += sum_col sigmoid(hp@Wb+bb)[row,col] * abuf[row,col] * wc[col]
__global__ __launch_bounds__(256) void gemm_attn_dot_kernel(
    const float* __restrict__ A, const float* __restrict__ B,
    const float* __restrict__ bias, const float* __restrict__ abuf,
    const float* __restrict__ wc, float* __restrict__ s,
    int M, int K, int Nn, int lda)
{
    __shared__ float  Araw[64][36];
    __shared__ bf16_t Bst[32][40];

    const int tid  = threadIdx.x;
    const int lane = tid & 31;
    const int wave = tid >> 5;
    const int wm = wave >> 1, wn = wave & 1;
    const int r16 = lane & 15, hsel = lane >> 4;
    const int m0 = blockIdx.y * 64;
    const int n0 = blockIdx.x * 32;
    const uint32_t a_lds = (uint32_t)(uintptr_t)(void*)&Araw[0][0];

    v8f acc = {0.f, 0.f, 0.f, 0.f, 0.f, 0.f, 0.f, 0.f};

    const int bk = tid >> 3;
    const int bc = (tid & 7) * 4;

    for (int k0 = 0; k0 < K; k0 += 32) {
        if (wave == 0)
            tdm_load_tile_f32(A + (size_t)m0 * lda + k0, a_lds, lda, M - m0);
        {
            const float4 v = *(const float4*)(B + (size_t)(k0 + bk) * Nn + n0 + bc);
            Bst[bc + 0][bk] = (bf16_t)v.x;
            Bst[bc + 1][bk] = (bf16_t)v.y;
            Bst[bc + 2][bk] = (bf16_t)v.z;
            Bst[bc + 3][bk] = (bf16_t)v.w;
        }
        if (wave == 0) __builtin_amdgcn_s_wait_tensorcnt(0);
        __syncthreads();

        v16bf af  = frag_a_from_f32(&Araw[wm * 16 + r16][0], hsel);
        v16bf bfv = frag_b_from_bf16(&Bst[wn * 16 + r16][0], hsel);
        acc = __builtin_amdgcn_wmma_f32_16x16x32_bf16(
            false, af, false, bfv, (short)0, acc, false, false);
        __syncthreads();
    }

    const int col = n0 + wn * 16 + r16;
    const float bv  = bias[col];
    const float wcv = wc[col];
#pragma unroll
    for (int i = 0; i < 8; ++i) {
        int row = m0 + wm * 16 + i + 8 * hsel;
        float part = 0.f;
        if (row < M) {
            float bsig = 1.f / (1.f + __expf(-(acc[i] + bv)));
            part = bsig * abuf[(size_t)row * 1024 + col] * wcv;
        }
        for (int off = 8; off >= 1; off >>= 1)   // reduce 16 cols of the tile
            part += __shfl_xor(part, off, 16);
        if (r16 == 0 && row < M) atomicAdd(&s[row], part);
    }
}

// ---------------- graph aggregation (GENConv softmax aggr) -----------------
__global__ void fill_int_kernel(int* __restrict__ p, int v, int n) {
    int i = blockIdx.x * blockDim.x + threadIdx.x;
    if (i < n) p[i] = v;
}

// order-preserving float->int key so atomicMax(int) == float max
__device__ __forceinline__ int f2key(float f) {
    int zi = __float_as_int(f);
    return (zi >= 0) ? zi : (zi ^ 0x7fffffff);
}
__device__ __forceinline__ float key2f(int k) {
    return __int_as_float((k >= 0) ? k : (k ^ 0x7fffffff));
}

__global__ __launch_bounds__(256) void edge_pass1(
    const float* __restrict__ x, const int* __restrict__ src,
    const int* __restrict__ dst, const float* __restrict__ tptr,
    int* __restrict__ zmaxI, int ldx)
{
    int e = blockIdx.x, ch = threadIdx.x;
    int sN = src[e], dN = dst[e];
    float t = tptr[0];
    float v = x[(size_t)sN * ldx + ch];
    float m = fmaxf(v, 0.f) + 1e-7f;
    atomicMax(&zmaxI[dN * 256 + ch], f2key(t * m));
}

__global__ __launch_bounds__(256) void edge_pass2(
    const float* __restrict__ x, const int* __restrict__ src,
    const int* __restrict__ dst, const float* __restrict__ tptr,
    const int* __restrict__ zmaxI, float* __restrict__ denom,
    float* __restrict__ aggnum, int ldx)
{
    int e = blockIdx.x, ch = threadIdx.x;
    int sN = src[e], dN = dst[e];
    float t = tptr[0];
    float v = x[(size_t)sN * ldx + ch];
    float m = fmaxf(v, 0.f) + 1e-7f;
    float z = t * m;
    float zm = key2f(zmaxI[dN * 256 + ch]);
    float ez = __expf(z - zm);
    atomicAdd(&denom[dN * 256 + ch], ez);
    atomicAdd(&aggnum[dN * 256 + ch], ez * m);
}

__global__ void combine_kernel(const float* __restrict__ x,
                               const float* __restrict__ denom,
                               const float* __restrict__ aggnum,
                               float* __restrict__ out, int ldx, int total)
{
    int i = blockIdx.x * blockDim.x + threadIdx.x;
    if (i >= total) return;
    int n = i >> 8, ch = i & 255;
    float den = denom[i];
    float agg = (den > 0.f) ? aggnum[i] / den : 0.f;   // isolated node -> 0
    out[i] = agg + x[(size_t)n * ldx + ch];            // GENConv root residual
}

__global__ void copy_layer0_kernel(const float* __restrict__ cv,
                                   float* __restrict__ feats, int total)
{
    int i = blockIdx.x * blockDim.x + threadIdx.x;
    if (i >= total) return;
    int n = i >> 8, ch = i & 255;
    feats[(size_t)n * 1024 + 256 + ch] = cv[i];
}

// ---------------- LayerNorm kernels ----------------------------------------
__global__ __launch_bounds__(256) void ln_relu512_kernel(
    const float* __restrict__ y, const float* __restrict__ w,
    const float* __restrict__ b, float* __restrict__ out)
{
    __shared__ float red[256];
    int n = blockIdx.x, t = threadIdx.x;
    const float* row = y + (size_t)n * 512;
    float v0 = row[t], v1 = row[t + 256];
    red[t] = v0 + v1; __syncthreads();
    for (int off = 128; off > 0; off >>= 1) { if (t < off) red[t] += red[t + off]; __syncthreads(); }
    float mu = red[0] * (1.f / 512.f); __syncthreads();
    float d0 = v0 - mu, d1 = v1 - mu;
    red[t] = d0 * d0 + d1 * d1; __syncthreads();
    for (int off = 128; off > 0; off >>= 1) { if (t < off) red[t] += red[t + off]; __syncthreads(); }
    float inv = rsqrtf(red[0] * (1.f / 512.f) + 1e-5f);
    out[(size_t)n * 512 + t]       = fmaxf(d0 * inv * w[t] + b[t], 0.f);
    out[(size_t)n * 512 + t + 256] = fmaxf(d1 * inv * w[t + 256] + b[t + 256], 0.f);
}

// h_next = h_prev + relu(LN256(cv)); feats slices in/out
__global__ __launch_bounds__(256) void layer_combine_kernel(
    const float* __restrict__ cv, const float* __restrict__ w,
    const float* __restrict__ b, float* __restrict__ feats,
    int prevoff, int outoff)
{
    __shared__ float red[256];
    int n = blockIdx.x, t = threadIdx.x;
    float v = cv[(size_t)n * 256 + t];
    red[t] = v; __syncthreads();
    for (int off = 128; off > 0; off >>= 1) { if (t < off) red[t] += red[t + off]; __syncthreads(); }
    float mu = red[0] * (1.f / 256.f); __syncthreads();
    float d = v - mu;
    red[t] = d * d; __syncthreads();
    for (int off = 128; off > 0; off >>= 1) { if (t < off) red[t] += red[t + off]; __syncthreads(); }
    float inv = rsqrtf(red[0] * (1.f / 256.f) + 1e-5f);
    float r = fmaxf(d * inv * w[t] + b[t], 0.f);
    feats[(size_t)n * 1024 + outoff + t] =
        feats[(size_t)n * 1024 + prevoff + t] + r;
}

// ---------------- attention head -------------------------------------------
__global__ __launch_bounds__(1024) void softmax_kernel(
    const float* __restrict__ s, float* __restrict__ A, int n)
{
    __shared__ float red[1024];
    int t = threadIdx.x;
    float m = -3.4e38f;
    for (int i = t; i < n; i += 1024) m = fmaxf(m, s[i]);
    red[t] = m; __syncthreads();
    for (int off = 512; off > 0; off >>= 1) { if (t < off) red[t] = fmaxf(red[t], red[t + off]); __syncthreads(); }
    float mx = red[0]; __syncthreads();
    float sum = 0.f;
    for (int i = t; i < n; i += 1024) sum += __expf(s[i] - mx);
    red[t] = sum; __syncthreads();
    for (int off = 512; off > 0; off >>= 1) { if (t < off) red[t] += red[t + off]; __syncthreads(); }
    float inv = 1.f / red[0];
    for (int i = t; i < n; i += 1024) A[i] = __expf(s[i] - mx) * inv;
}

__global__ __launch_bounds__(1024) void pool_kernel(
    const float* __restrict__ A, const float* __restrict__ hp,
    float* __restrict__ hpool, int n)
{
    int k = threadIdx.x;
    int chunk = (n + gridDim.x - 1) / gridDim.x;
    int n0 = blockIdx.x * chunk;
    int n1 = min(n, n0 + chunk);
    float acc = 0.f;
    for (int i = n0; i < n1; ++i) acc += A[i] * hp[(size_t)i * 1024 + k];
    atomicAdd(&hpool[k], acc);
}

__global__ __launch_bounds__(1024) void head_kernel(
    const float* __restrict__ hpool, const float* __restrict__ rho_w,
    const float* __restrict__ rho_b, const float* __restrict__ clf_w,
    const float* __restrict__ clf_b, float* __restrict__ out)
{
    __shared__ float h2[1024];
    __shared__ float r0[1024];
    __shared__ float r1[1024];
    int j = threadIdx.x;
    float acc = rho_b[j];
    for (int k = 0; k < 1024; ++k) acc += hpool[k] * rho_w[(size_t)k * 1024 + j];
    h2[j] = fmaxf(acc, 0.f);
    __syncthreads();
    r0[j] = h2[j] * clf_w[j * 2 + 0];
    r1[j] = h2[j] * clf_w[j * 2 + 1];
    __syncthreads();
    for (int off = 512; off > 0; off >>= 1) {
        if (j < off) { r0[j] += r0[j + off]; r1[j] += r1[j + off]; }
        __syncthreads();
    }
    if (j == 0) {
        float l0 = r0[0] + clf_b[0], l1 = r1[0] + clf_b[1];
        out[0] = l0;
        out[1] = l1;
        out[2] = (l1 > l0) ? 1.f : 0.f;   // argmax (first index on ties)
    }
}

// ---------------------------------------------------------------------------
extern "C" void kernel_launch(void* const* d_in, const int* in_sizes, int n_in,
                              void* d_out, int out_size, void* d_ws, size_t ws_size,
                              hipStream_t stream)
{
    const float* x     = (const float*)d_in[0];
    const int*   ei    = (const int*)d_in[1];
    const float* fc_w  = (const float*)d_in[2];
    const float* fc_b  = (const float*)d_in[3];
    const float* cw1   = (const float*)d_in[4];
    const float* cb1   = (const float*)d_in[5];
    const float* clnw  = (const float*)d_in[6];
    const float* clnb  = (const float*)d_in[7];
    const float* cw2   = (const float*)d_in[8];
    const float* cb2   = (const float*)d_in[9];
    const float* ct    = (const float*)d_in[10];
    const float* nw    = (const float*)d_in[11];
    const float* nb    = (const float*)d_in[12];
    const float* phi_w = (const float*)d_in[13];
    const float* phi_b = (const float*)d_in[14];
    const float* aw    = (const float*)d_in[15];
    const float* ab    = (const float*)d_in[16];
    const float* bw    = (const float*)d_in[17];
    const float* bb    = (const float*)d_in[18];
    const float* cwc   = (const float*)d_in[19];
    // d_in[20] = attn_c_b: constant shift before softmax -> no effect, unused
    const float* rho_w = (const float*)d_in[21];
    const float* rho_b = (const float*)d_in[22];
    const float* clf_w = (const float*)d_in[23];
    const float* clf_b = (const float*)d_in[24];

    const int N = in_sizes[0] / 768;
    const int E = in_sizes[1] / 2;
    const int* src = ei;
    const int* dst = ei + E;

    // workspace layout (floats); hp/abuf alias conv-phase scratch
    float* ws     = (float*)d_ws;
    float* feats  = ws;                               // N*1024
    float* convin = feats  + (size_t)N * 1024;        // N*256
    float* ybuf   = convin + (size_t)N * 256;         // N*512
    float* midbuf = ybuf   + (size_t)N * 512;         // N*512
    float* cvbuf  = midbuf + (size_t)N * 512;         // N*256
    int*   zmaxI  = (int*)(cvbuf + (size_t)N * 256);  // N*256
    float* denom  = (float*)(zmaxI + (size_t)N * 256);// N*256
    float* aggnum = denom  + (size_t)N * 256;         // N*256
    float* sbuf   = aggnum + (size_t)N * 256;         // N
    float* hpool  = sbuf + N;                         // 1024
    float* hp     = cvbuf;    // alias: N*1024 over [cv..aggnum]
    float* abuf   = convin;   // alias: N*1024 over [convin..mid]
    float* Aout   = (float*)d_out + 3;                // attention weights [N]

    dim3 blk(256);
    auto gemmGrid = [](int M, int Nn) {
        return dim3((unsigned)(Nn / 32), (unsigned)((M + 63) / 64));
    };

    // fc: h0 = relu(x @ fc_w + fc_b) -> feats[:, 0:256]
    gemm_kernel<ACT_RELU><<<gemmGrid(N, 256), blk, 0, stream>>>(
        x, fc_w, fc_b, feats, N, 768, 256, 768, 1024);

    for (int L = 0; L < 3; ++L) {
        const float* xin = feats + L * 256;   // row stride 1024
        fill_int_kernel<<<(N * 256 + 1023) / 1024, 1024, 0, stream>>>(
            zmaxI, (int)0x80000000, N * 256);
        (void)hipMemsetAsync(denom, 0, (size_t)N * 256 * sizeof(float), stream);
        (void)hipMemsetAsync(aggnum, 0, (size_t)N * 256 * sizeof(float), stream);

        edge_pass1<<<E, 256, 0, stream>>>(xin, src, dst, ct + L, zmaxI, 1024);
        edge_pass2<<<E, 256, 0, stream>>>(xin, src, dst, ct + L, zmaxI,
                                          denom, aggnum, 1024);
        combine_kernel<<<(N * 256 + 255) / 256, 256, 0, stream>>>(
            xin, denom, aggnum, convin, 1024, N * 256);

        // MLP: Lin(256,512) -> LN -> ReLU -> Lin(512,256)
        gemm_kernel<ACT_NONE><<<gemmGrid(N, 512), blk, 0, stream>>>(
            convin, cw1 + (size_t)L * 256 * 512, cb1 + L * 512, ybuf,
            N, 256, 512, 256, 512);
        ln_relu512_kernel<<<N, 256, 0, stream>>>(
            ybuf, clnw + L * 512, clnb + L * 512, midbuf);
        gemm_kernel<ACT_NONE><<<gemmGrid(N, 256), blk, 0, stream>>>(
            midbuf, cw2 + (size_t)L * 512 * 256, cb2 + L * 256, cvbuf,
            N, 512, 256, 512, 256);

        if (L == 0)
            copy_layer0_kernel<<<(N * 256 + 255) / 256, 256, 0, stream>>>(
                cvbuf, feats, N * 256);
        else
            layer_combine_kernel<<<N, 256, 0, stream>>>(
                cvbuf, nw + L * 256, nb + L * 256, feats,
                L * 256, (L + 1) * 256);
    }

    // hp = relu(feats @ phi_w + phi_b)
    gemm_kernel<ACT_RELU><<<gemmGrid(N, 1024), blk, 0, stream>>>(
        feats, phi_w, phi_b, hp, N, 1024, 1024, 1024, 1024);
    // a = tanh(hp @ Wa + ba)
    gemm_kernel<ACT_TANH><<<gemmGrid(N, 1024), blk, 0, stream>>>(
        hp, aw, ab, abuf, N, 1024, 1024, 1024, 1024);
    // s[n] = sum_k sigmoid(hp@Wb+bb) * a * wc  (cb dropped: softmax-invariant)
    (void)hipMemsetAsync(sbuf, 0, (size_t)N * sizeof(float), stream);
    gemm_attn_dot_kernel<<<gemmGrid(N, 1024), blk, 0, stream>>>(
        hp, bw, bb, abuf, cwc, sbuf, N, 1024, 1024, 1024);

    softmax_kernel<<<1, 1024, 0, stream>>>(sbuf, Aout, N);

    (void)hipMemsetAsync(hpool, 0, 1024 * sizeof(float), stream);
    pool_kernel<<<128, 1024, 0, stream>>>(Aout, hp, hpool, N);

    head_kernel<<<1, 1024, 0, stream>>>(hpool, rho_w, rho_b, clf_w, clf_b,
                                        (float*)d_out);
}
